// NetworkLocal_26757646254249
// MI455X (gfx1250) — matrix operand order repaired and checked by reading, not compiled, not run on hardware
//
#include <hip/hip_runtime.h>
#include <hip/hip_bf16.h>
#include <math.h>

// ---------------------------------------------------------------------------
// Fused feature + 3-layer MLP + pool for MI455X (gfx1250), f16 WMMA path.
//   B=4, T=256, D=512, K=3, k_in=17(->32), H=196(->224), k_latent=128
//   Grid 512 WGs, each handles 2 (b,t) tiles. 256 threads = 8 wave32.
//   Layer3 folded through the (linear) pooling: out = (sum_d h2) @ W2 + D*b2.
//   Two-tile interleaving hides the WMMA->VALU hazard window (4 coexec NOPs).
// ---------------------------------------------------------------------------

typedef __attribute__((ext_vector_type(16))) _Float16 v16h;
typedef __attribute__((ext_vector_type(8)))  _Float16 v8h;
typedef __attribute__((ext_vector_type(8)))  float    v8f;
typedef __attribute__((ext_vector_type(4)))  float    v4f;

#define EPSF 1e-5f
#define HP   224      // padded hidden (196 -> 224 = 7*32)
#define KINP 32       // padded k_in (17 -> 32)
#define CH   128      // rows per chunk

__device__ __forceinline__ v16h cat16(v8h lo, v8h hi) {
  return __builtin_shufflevector(lo, hi, 0,1,2,3,4,5,6,7,8,9,10,11,12,13,14,15);
}

// A-fragment (16x32 f16, ISA 7.12.2): lane&15 -> M row; halves at K = kg..kg+7
// and kg+16..kg+23 with kg = kb + (lane>=16 ? 8 : 0). Contiguous 16B runs.
__device__ __forceinline__ v16h load_a_frag(const _Float16* p, int strideH,
                                            int mb, int kb, int lane) {
  int row = mb + (lane & 15);
  int kg  = kb + ((lane >> 4) << 3);
  const _Float16* base = p + row * strideH + kg;
  v8h lo = *(const v8h*)(base);
  v8h hi = *(const v8h*)(base + 16);
  return cat16(lo, hi);
}

// B-fragment (32x16 f16) from a [col][k] row-major array:
//   col=nb+(lane&15); halves at K = kg..kg+7 and kg+16..kg+23, kg = kb + (lane>=16 ? 8 : 0).
__device__ __forceinline__ v16h load_b_frag(const _Float16* wt, int strideK,
                                            int nb, int kb, int lane) {
  int col = nb + (lane & 15);
  int kg  = kb + ((lane >> 4) << 3);
  const _Float16* base = wt + col * strideK + kg;
  v8h lo = *(const v8h*)(base);
  v8h hi = *(const v8h*)(base + 16);
  return cat16(lo, hi);
}

__device__ __forceinline__ v8f wmma_f16(v16h a, v16h b, v8f c) {
  return __builtin_amdgcn_wmma_f32_16x16x32_f16(false, a, false, b,
                                                (short)0, c, false, false);
}

__device__ __forceinline__ v8h lrelu_pack(v8f acc) {
  v8h hv;
  #pragma unroll
  for (int r = 0; r < 8; ++r) {
    float f = acc[r];
    f = fmaxf(f, 0.01f * f);                    // leaky_relu, slope<1
    hv[r] = (_Float16)f;
  }
  return hv;
}

__global__ __launch_bounds__(256)
void fused_mlp_pool_kernel(const float* __restrict__ x0,
                           const float* __restrict__ x,
                           const int*   __restrict__ Nc,
                           const float* __restrict__ basis,
                           const float* __restrict__ v,
                           const float* __restrict__ W0,
                           const float* __restrict__ b0,
                           const float* __restrict__ W1,
                           const float* __restrict__ b1,
                           const float* __restrict__ W2,
                           const float* __restrict__ b2,
                           float* __restrict__ out)
{
  extern __shared__ __align__(16) char smem[];
  _Float16* sW1t = (_Float16*)smem;            // [224][224] = 100352 B  (W1^T)
  _Float16* sW0t = sW1t + HP * HP;             // [224][32]  =  14336 B  (W0^T, K-permuted)
  _Float16* sF   = sW0t + HP * KINP;           // [128][32]  =   8192 B  (features, K-permuted)
  _Float16* sH1  = sF   + CH * KINP;           // [128][224] =  57344 B
  float*    sB0  = (float*)(sH1 + CH * HP);    // [224]
  float*    sB1  = sB0 + HP;                   // [224]
  float*    sPool= sB1 + HP;                   // [224] pooled layer-2 activations
  float*    sVec = sPool + HP;                 // x0n[3], bn[9]

  const int tid  = threadIdx.x;
  const int lane = tid & 31;
  const int wv   = tid >> 5;      // wave id 0..7
  const int hi8  = (lane >> 4) << 3;

  // ---- stage weights once per WG (f32 -> f16, transposed, zero-padded) ----
  // Layer-1 K-permutation: slot k<16 -> original feature k+1, slot 16 -> Nf.
  for (int i = tid; i < HP * HP; i += 256) {
    int n = i / HP, k = i - n * HP;
    float val = (n < 196 && k < 196) ? W1[k * 196 + n] : 0.f;
    sW1t[i] = (_Float16)val;
  }
  for (int i = tid; i < HP * KINP; i += 256) {
    int n = i >> 5, k = i & 31;
    float val = 0.f;
    if (n < 196) {
      if (k < 16)       val = W0[(k + 1) * 196 + n];
      else if (k == 16) val = W0[n];              // Nf row (orig k=0)
    }
    sW0t[i] = (_Float16)val;
  }
  for (int i = tid; i < HP; i += 256) {
    sB0[i] = (i < 196) ? b0[i] : 0.f;
    sB1[i] = (i < 196) ? b1[i] : 0.f;
  }
  __syncthreads();

  for (int rep = 0; rep < 2; ++rep) {
    const int bt = blockIdx.x * 2 + rep;   // b*256 + t
    const int b  = bt >> 8;

    // per-(b,t) setup
    for (int i = tid; i < HP; i += 256) sPool[i] = 0.f;
    if (tid == 0) {
      float a0 = x0[bt*3+0], a1 = x0[bt*3+1], a2 = x0[bt*3+2];
      float inv = 1.f / (sqrtf(a0*a0 + a1*a1 + a2*a2) + EPSF);
      sVec[0] = a0*inv; sVec[1] = a1*inv; sVec[2] = a2*inv;
      for (int k = 0; k < 3; ++k) {
        const float* bp = basis + ((size_t)b*3 + k)*3;
        float c0 = bp[0], c1 = bp[1], c2 = bp[2];
        float bi = 1.f / (sqrtf(c0*c0 + c1*c1 + c2*c2) + EPSF);
        sVec[3 + k*3 + 0] = c0*bi;
        sVec[3 + k*3 + 1] = c1*bi;
        sVec[3 + k*3 + 2] = c2*bi;
      }
    }
    int nv = Nc[bt];
    float Nf = (float)(nv > 1 ? nv : 1);
    __syncthreads();

    for (int chunk = 0; chunk < 4; ++chunk) {
      // ---- features: 2 threads per row (x-family / v-family), b128 stores --
      {
        int r = tid >> 1, s = tid & 1;
        int d = chunk * CH + r;
        const float* up = (s ? v : x) + (((size_t)bt * 512) + d) * 3;
        float u0 = up[0], u1 = up[1], u2 = up[2];
        float mag = sqrtf(u0*u0 + u1*u1 + u2*u2);
        float inv = 1.f / (mag + EPSF);
        float h0 = u0*inv, h1 = u1*inv, h2 = u2*inv;
        v8h fam;
        fam[0] = (_Float16)mag;
        fam[1] = (_Float16)(h0*sVec[0] + h1*sVec[1] + h2*sVec[2]);
        #pragma unroll
        for (int k = 0; k < 3; ++k) {
          float bx = sVec[3+k*3+0], by = sVec[3+k*3+1], bz = sVec[3+k*3+2];
          fam[2+k] = (_Float16)(h0*bx + h1*by + h2*bz);
          fam[5+k] = (_Float16)(u0*bx + u1*by + u2*bz);
        }
        _Float16* fr = sF + r * KINP;
        *(v8h*)(fr + s * 8) = fam;              // slots [0..7] or [8..15]
        v8h tail = {0,0,0,0,0,0,0,0};
        if (s == 0) tail[0] = (_Float16)Nf;     // slot 16 = Nf, 17..23 pad
        *(v8h*)(fr + 16 + s * 8) = tail;        // slots [16..23] / [24..31]
      }
      __syncthreads();

      // ---- layer 1 (swapped roles): h1^T = W0^T x F^T, two m-tiles at once -
      {
        const int nb  = wv * 16;
        const int row = nb + (lane & 15);
        v16h bf = load_b_frag(sF, KINP, nb, 0, lane);   // reused for all m
        for (int m2 = 0; m2 < 7; ++m2) {
          int m0 = m2 * 32, m1 = m0 + 16;
          // issue all loads for both tiles first
          v16h a0 = load_a_frag(sW0t, KINP, m0, 0, lane);
          v16h a1 = load_a_frag(sW0t, KINP, m1, 0, lane);
          v4f b0lo = *(const v4f*)(sB0 + m0 + hi8);
          v4f b0hi = *(const v4f*)(sB0 + m0 + hi8 + 4);
          v4f b1lo = *(const v4f*)(sB0 + m1 + hi8);
          v4f b1hi = *(const v4f*)(sB0 + m1 + hi8 + 4);
          v8f acc0 = {b0lo[0], b0lo[1], b0lo[2], b0lo[3],
                      b0hi[0], b0hi[1], b0hi[2], b0hi[3]};
          v8f acc1 = {b1lo[0], b1lo[1], b1lo[2], b1lo[3],
                      b1hi[0], b1hi[1], b1hi[2], b1hi[3]};
          // back-to-back independent WMMAs (no D->A/B overlap => no NOPs)
          acc0 = wmma_f16(a0, bf, acc0);
          acc1 = wmma_f16(a1, bf, acc1);
          // epilogues fill the hazard windows
          v8h hv0 = lrelu_pack(acc0);
          v8h hv1 = lrelu_pack(acc1);
          *(v8h*)(&sH1[row * HP + m0 + hi8]) = hv0;
          *(v8h*)(&sH1[row * HP + m1 + hi8]) = hv1;
        }
      }
      __syncthreads();

      // ---- layer 2: h1[128,224] @ W1[224,224], two n-tiles at once ---------
      {
        const int mb = wv * 16;
        v16h af[7];
        #pragma unroll
        for (int kk = 0; kk < 7; ++kk)
          af[kk] = load_a_frag(sH1, HP, mb, kk * 32, lane);

        for (int n2 = 0; n2 < 7; ++n2) {
          int nb0 = n2 * 32, nb1 = nb0 + 16;
          int col0 = nb0 + (lane & 15), col1 = nb1 + (lane & 15);
          float bv0 = sB1[col0], bv1 = sB1[col1];
          v8f acc0 = {bv0, bv0, bv0, bv0, bv0, bv0, bv0, bv0};
          v8f acc1 = {bv1, bv1, bv1, bv1, bv1, bv1, bv1, bv1};
          #pragma unroll
          for (int kk = 0; kk < 7; ++kk) {
            v16h bf0 = load_b_frag(sW1t, HP, nb0, kk * 32, lane);
            v16h bf1 = load_b_frag(sW1t, HP, nb1, kk * 32, lane);
            acc0 = wmma_f16(af[kk], bf0, acc0);   // alternating accumulators:
            acc1 = wmma_f16(af[kk], bf1, acc1);   // consecutive WMMAs independent
          }
          // lrelu then column-sum the 16-row tiles; pooling over rows is
          // linear, so layer 3 only needs sum_d h2[d,:].
          float s0 = 0.f, s1 = 0.f;
          #pragma unroll
          for (int r = 0; r < 8; ++r) {
            s0 += fmaxf(acc0[r], 0.01f * acc0[r]);
            s1 += fmaxf(acc1[r], 0.01f * acc1[r]);
          }
          atomicAdd(&sPool[col0], s0);            // ds_add_f32
          atomicAdd(&sPool[col1], s1);
        }
      }
      __syncthreads();
    }

    // ---- layer 3 on pooled vector: (sum h2) @ W2 + 512*b2, divide by Nf ----
    if (tid < 128) {
      float acc = 512.f * b2[tid];
      #pragma unroll 4
      for (int k = 0; k < 196; ++k)
        acc = fmaf(sPool[k], W2[k * 128 + tid], acc);
      out[(size_t)bt * 128 + tid] = acc / Nf;
    }
    __syncthreads();   // protect sPool/sVec reuse in next rep
  }
}

extern "C" void kernel_launch(void* const* d_in, const int* in_sizes, int n_in,
                              void* d_out, int out_size, void* d_ws, size_t ws_size,
                              hipStream_t stream) {
  const float* x0    = (const float*)d_in[0];
  const float* x     = (const float*)d_in[1];
  const int*   N     = (const int*)  d_in[2];
  const float* basis = (const float*)d_in[3];
  const float* v     = (const float*)d_in[4];
  const float* W0    = (const float*)d_in[5];
  const float* b0    = (const float*)d_in[6];
  const float* W1    = (const float*)d_in[7];
  const float* b1    = (const float*)d_in[8];
  const float* W2    = (const float*)d_in[9];
  const float* b2    = (const float*)d_in[10];
  float* out = (float*)d_out;

  const size_t smem =
      (size_t)(HP*HP + HP*KINP + CH*KINP + CH*HP) * sizeof(_Float16) +
      (size_t)(3*HP + 16) * sizeof(float);   // ~183 KB, within 320 KB WGP LDS

  hipFuncSetAttribute((const void*)fused_mlp_pool_kernel,
                      hipFuncAttributeMaxDynamicSharedMemorySize, (int)smem);

  fused_mlp_pool_kernel<<<512, 256, smem, stream>>>(
      x0, x, N, basis, v, W0, b0, W1, b1, W2, b2, out);
}
